// SwapNumPredictHeteroMix_57312043597873
// MI455X (gfx1250) — compile-verified, matrix operand order checked by reading, not run on hardware
//
#include <hip/hip_runtime.h>

// ---------------------------------------------------------------------------
// CDNA5 (gfx1250) implementation of the GAT + conv/MLP head pipeline.
// wave32 everywhere. Dense matmuls use v_wmma_f32_16x16x32_f16 with
// pre-padded f16 activations (branch-free b128 fragment loads), a fully
// unrolled K loop, and compile-time strides so the C-tile store uses one
// base address + immediate offsets.
// ---------------------------------------------------------------------------

#define NN 32768
#define DEG 16
#define NPGC 1024
#define BGRP 32
#define KTOP 30
#define GCO 32

typedef __attribute__((ext_vector_type(16))) _Float16 v16h;
typedef __attribute__((ext_vector_type(8)))  _Float16 v8h;
typedef __attribute__((ext_vector_type(8)))  float    v8f;

// K index inside a 16x32 f16 WMMA fragment for a given lane & element.
// halfs [0..7] are K = kb..kb+7 and halfs [8..15] are K = kb+16..kb+23,
// kb = 0 for lanes 0-15, 8 for lanes 16-31 (ISA 7.12.2).
__device__ __forceinline__ int frag_k(int lane, int e) {
  int r = e >> 1, p = e & 1;
  int base = (lane < 16) ? 0 : 8;
  int off = (r < 4) ? (2 * r) : (16 + 2 * (r - 4));
  return base + off + p;
}

// ------------------- activation cast + zero-pad (f32 -> f16) ---------------
__global__ void cast_pad(const float* __restrict__ h, _Float16* __restrict__ o,
                         int fi, int fi_pad) {
  int row = blockIdx.x;
  int f = threadIdx.x * 2;
  const float* r = h + (size_t)row * fi;
  union { _Float16 hh[2]; unsigned u; } pk;
  pk.hh[0] = (f < fi) ? (_Float16)r[f] : (_Float16)0.f;
  pk.hh[1] = (f + 1 < fi) ? (_Float16)r[f + 1] : (_Float16)0.f;
  *reinterpret_cast<unsigned*>(o + (size_t)row * fi_pad + f) = pk.u;
}

// --------------------------- weight repack ---------------------------------
__global__ void repack_w(const float* __restrict__ W, _Float16* __restrict__ P,
                         int fi, int fo, int nkt, int nct) {
  int idx = blockIdx.x * blockDim.x + threadIdx.x;
  int total = nct * nkt * 32;
  if (idx >= total) return;
  int lane = idx & 31;
  int t = idx >> 5;
  int kt = t % nkt;
  int ct = t / nkt;
  int n = ct * 16 + (lane & 15);
  _Float16* out = P + (size_t)idx * 16;
#pragma unroll
  for (int e = 0; e < 16; ++e) {
    int k = kt * 32 + frag_k(lane, e);
    float v = (k < fi && n < fo) ? W[(size_t)k * fo + n] : 0.f;
    out[e] = (_Float16)v;
  }
}

// --------------------------- WMMA GEMM -------------------------------------
template <int NKT, int FOP>
__global__ void gemm_wmma(const _Float16* __restrict__ A16,
                          const _Float16* __restrict__ P,
                          float* __restrict__ C) {
  constexpr int FIP = NKT * 32;
  int wave = threadIdx.x >> 5;
  int lane = threadIdx.x & 31;
  int rowTile = blockIdx.x * 8 + wave;
  int ct = blockIdx.y;
  int m0 = rowTile * 16;
  int m = m0 + (lane & 15);
  int kb = (lane < 16) ? 0 : 8;
  const _Float16* Arow = A16 + (size_t)m * FIP + kb;
  const _Float16* Pb = P + (((size_t)ct * NKT) * 32 + (size_t)lane) * 16;

  v8f acc = {};
#pragma unroll
  for (int kt = 0; kt < NKT; ++kt) {
    v8h lo = *reinterpret_cast<const v8h*>(Arow + kt * 32);
    v8h hi = *reinterpret_cast<const v8h*>(Arow + kt * 32 + 16);
    v16h a;
#pragma unroll
    for (int e = 0; e < 8; ++e) { a[e] = lo[e]; a[8 + e] = hi[e]; }
    v16h b = *reinterpret_cast<const v16h*>(Pb + (size_t)kt * 32 * 16);
    acc = __builtin_amdgcn_wmma_f32_16x16x32_f16(
        false, a, false, b, (short)0, acc, false, false);
  }
  int col = ct * 16 + (lane & 15);
  int rbase = m0 + ((lane < 16) ? 0 : 8);
  float* Cout = C + (size_t)rbase * FOP + col;
#pragma unroll
  for (int v = 0; v < 8; ++v)
    Cout[v * FOP] = acc[v];   // immediate-offset stores (FOP constexpr)
}

// ----------------- attention coefficients (one wave per node) --------------
__global__ void attn_dots(const float* __restrict__ hW,
                          const float* __restrict__ a_src,
                          const float* __restrict__ a_dst,
                          float* __restrict__ as_, float* __restrict__ ad_,
                          int fo, int fo_pad) {
  int lane = threadIdx.x & 31;
  int node = (blockIdx.x * blockDim.x + threadIdx.x) >> 5;
  const float* r = hW + (size_t)node * fo_pad;
  float s1 = 0.f, s2 = 0.f;
  for (int f = lane; f < fo; f += 32) {
    float v = r[f];
    s1 += v * a_src[f];
    s2 += v * a_dst[f];
  }
#pragma unroll
  for (int o = 16; o > 0; o >>= 1) {
    s1 += __shfl_xor(s1, o, 32);
    s2 += __shfl_xor(s2, o, 32);
  }
  if (lane == 0) { as_[node] = s1; ad_[node] = s2; }
}

// --------------------- GAT aggregation (one wave per node) -----------------
__global__ void gat_aggregate(const int* __restrict__ esrc,
                              const float* __restrict__ hW,
                              const float* __restrict__ hWlin,
                              const float* __restrict__ as_,
                              const float* __restrict__ ad_,
                              const float* __restrict__ gb,
                              const float* __restrict__ lb,
                              float* __restrict__ out,
                              int fo, int fo_pad, int out_stride, int last) {
  int lane = threadIdx.x & 31;
  int node = (blockIdx.x * blockDim.x + threadIdx.x) >> 5;

  int s = node;
  if (lane < DEG) s = esrc[node * DEG + lane];

  float e;
  if (lane <= DEG) {
    float t = as_[s] + ad_[node];
    e = (t > 0.f) ? t : 0.2f * t;   // leaky_relu NEG_GAT
  } else {
    e = -3.0e38f;
  }
  float m = e;
#pragma unroll
  for (int o = 16; o > 0; o >>= 1) m = fmaxf(m, __shfl_xor(m, o, 32));
  float w = (lane <= DEG) ? expf(e - m) : 0.f;
  float ssum = w;
#pragma unroll
  for (int o = 16; o > 0; o >>= 1) ssum += __shfl_xor(ssum, o, 32);
  float alpha = w / ssum;

  float al[DEG + 1];
  int sr[DEG + 1];
#pragma unroll
  for (int k = 0; k <= DEG; ++k) {
    al[k] = __shfl(alpha, k, 32);
    sr[k] = __shfl(s, k, 32);
  }

  // get the 17 scattered rows in flight early (global_prefetch_b8)
  if (lane <= DEG)
    __builtin_prefetch(&hW[(size_t)s * fo_pad], 0, 0);

  for (int f = lane; f < fo; f += 32) {
    float acc = 0.f;
#pragma unroll
    for (int k = 0; k <= DEG; ++k)
      acc += al[k] * hW[(size_t)sr[k] * fo_pad + f];
    acc += gb[f];
    float r;
    if (last) {
      r = acc;
    } else {
      r = (acc > 0.f) ? acc : 0.01f * acc;               // leaky NEG
      r += hWlin[(size_t)node * fo_pad + f] + lb[f];     // residual linear
    }
    out[(size_t)node * out_stride + f] = r;
  }
}

// ------------------------- top-K select + gather ---------------------------
__global__ void topk_gather(const float* __restrict__ h7, float* __restrict__ z) {
  int b = blockIdx.x, t = threadIdx.x;  // 256 threads
  __shared__ float sval[NPGC];
  __shared__ int staken[NPGC];
  __shared__ float rv[256];
  __shared__ int ri[256];
  __shared__ int order[KTOP];

  for (int j = t; j < NPGC; j += 256) {
    sval[j] = h7[(size_t)(b * NPGC + j) * GCO + (GCO - 1)];
    staken[j] = 0;
  }
  __syncthreads();

  for (int k = 0; k < KTOP; ++k) {
    float bv = -3.0e38f;
    int bi = 1 << 30;
    for (int j = t; j < NPGC; j += 256) {
      if (!staken[j]) {
        float v = sval[j];
        if (v > bv || (v == bv && j < bi)) { bv = v; bi = j; }
      }
    }
    rv[t] = bv; ri[t] = bi;
    __syncthreads();
    for (int off = 128; off > 0; off >>= 1) {
      if (t < off) {
        float ov = rv[t + off]; int oi = ri[t + off];
        if (ov > rv[t] || (ov == rv[t] && oi < ri[t])) { rv[t] = ov; ri[t] = oi; }
      }
      __syncthreads();
    }
    if (t == 0) { order[k] = ri[0]; staken[ri[0]] = 1; }
    __syncthreads();
  }

  for (int idx = t; idx < KTOP * GCO; idx += 256) {
    int k = idx >> 5, c = idx & 31;
    z[(size_t)b * (KTOP * GCO) + idx] =
        h7[(size_t)(b * NPGC + order[k]) * GCO + c];
  }
}

// ------------------------------ layernorm ----------------------------------
__global__ void layernorm_k(const float* __restrict__ z, const float* __restrict__ g,
                            const float* __restrict__ bb, float* __restrict__ out,
                            int D) {
  int b = blockIdx.x, t = threadIdx.x;
  __shared__ float red[256];
  const float* r = z + (size_t)b * D;
  float s = 0.f;
  for (int j = t; j < D; j += 256) s += r[j];
  red[t] = s; __syncthreads();
  for (int o = 128; o > 0; o >>= 1) { if (t < o) red[t] += red[t + o]; __syncthreads(); }
  float mu = red[0] / (float)D;
  __syncthreads();
  float v = 0.f;
  for (int j = t; j < D; j += 256) { float d = r[j] - mu; v += d * d; }
  red[t] = v; __syncthreads();
  for (int o = 128; o > 0; o >>= 1) { if (t < o) red[t] += red[t + o]; __syncthreads(); }
  float rs = rsqrtf(red[0] / (float)D + 1e-5f);
  float* o_ = out + (size_t)b * D;
  for (int j = t; j < D; j += 256) o_[j] = (r[j] - mu) * rs * g[j] + bb[j];
}

// ------------------- conv1 -> pool -> conv2 -> ln2 -> MLP ------------------
__global__ void head_k(const float* __restrict__ zn,
                       const float* __restrict__ c1w, const float* __restrict__ c1b,
                       const float* __restrict__ c2w, const float* __restrict__ c2b,
                       const float* __restrict__ g2, const float* __restrict__ b2,
                       const float* __restrict__ m1w, const float* __restrict__ m1b,
                       const float* __restrict__ m2w, const float* __restrict__ m2b,
                       const float* __restrict__ m3w, const float* __restrict__ m3b,
                       float* __restrict__ out) {
  int b = blockIdx.x, t = threadIdx.x;  // 256 threads
  __shared__ float zin[KTOP * GCO];     // 960
  __shared__ float z1[16 * 30];
  __shared__ float z1p[16 * 15];
  __shared__ float z2[352];
  __shared__ float red[256];
  __shared__ float m1o[100];
  __shared__ float m2o[30];

  for (int j = t; j < 960; j += 256) zin[j] = zn[(size_t)b * 960 + j];
  __syncthreads();

  for (int idx = t; idx < 16 * 30; idx += 256) {
    int c = idx / 30, tt = idx % 30;
    float a = c1b[c];
    for (int j = 0; j < 32; ++j) a += zin[tt * 32 + j] * c1w[c * 32 + j];
    z1[c * 30 + tt] = (a > 0.f) ? a : 0.01f * a;
  }
  __syncthreads();

  for (int idx = t; idx < 16 * 15; idx += 256) {
    int c = idx / 15, u = idx % 15;
    z1p[c * 15 + u] = fmaxf(z1[c * 30 + 2 * u], z1[c * 30 + 2 * u + 1]);
  }
  __syncthreads();

  for (int idx = t; idx < 352; idx += 256) {
    int o2 = idx / 11, tt = idx % 11;
    float a = c2b[o2];
    for (int c = 0; c < 16; ++c)
      for (int j = 0; j < 5; ++j)
        a += z1p[c * 15 + tt + j] * c2w[o2 * 80 + c * 5 + j];
    z2[idx] = a;
  }
  __syncthreads();

  float s = 0.f;
  for (int j = t; j < 352; j += 256) s += z2[j];
  red[t] = s; __syncthreads();
  for (int o = 128; o > 0; o >>= 1) { if (t < o) red[t] += red[t + o]; __syncthreads(); }
  float mu = red[0] / 352.f;
  __syncthreads();
  float v = 0.f;
  for (int j = t; j < 352; j += 256) { float d = z2[j] - mu; v += d * d; }
  red[t] = v; __syncthreads();
  for (int o = 128; o > 0; o >>= 1) { if (t < o) red[t] += red[t + o]; __syncthreads(); }
  float rs = rsqrtf(red[0] / 352.f + 1e-5f);
  __syncthreads();
  for (int j = t; j < 352; j += 256) z2[j] = (z2[j] - mu) * rs * g2[j] + b2[j];
  __syncthreads();

  for (int j = t; j < 100; j += 256) {
    float a = m1b[j];
    for (int i2 = 0; i2 < 352; ++i2) a += z2[i2] * m1w[i2 * 100 + j];
    m1o[j] = (a > 0.f) ? a : 0.01f * a;
  }
  __syncthreads();
  for (int j = t; j < 30; j += 256) {
    float a = m2b[j];
    for (int i2 = 0; i2 < 100; ++i2) a += m1o[i2] * m2w[i2 * 30 + j];
    m2o[j] = (a > 0.f) ? a : 0.01f * a;
  }
  __syncthreads();
  if (t == 0) {
    float a = m3b[0];
    for (int i2 = 0; i2 < 30; ++i2) a += m2o[i2] * m3w[i2];
    out[b] = a;
  }
}

// ---------------------------------------------------------------------------
extern "C" void kernel_launch(void* const* d_in, const int* in_sizes, int n_in,
                              void* d_out, int out_size, void* d_ws, size_t ws_size,
                              hipStream_t stream) {
  (void)in_sizes; (void)n_in; (void)out_size; (void)ws_size;

  const float* x    = (const float*)d_in[0];
  const int*   esrc = (const int*)d_in[1];   // row 0 of edge_index = src

  const float* gatW[7];  const float* gatAS[7]; const float* gatAD[7]; const float* gatB[7];
  const float* linW[6];  const float* linB[6];
  for (int i = 0; i < 7; ++i) {
    gatW[i]  = (const float*)d_in[3 + 4 * i];
    gatAS[i] = (const float*)d_in[4 + 4 * i];
    gatAD[i] = (const float*)d_in[5 + 4 * i];
    gatB[i]  = (const float*)d_in[6 + 4 * i];
  }
  for (int i = 0; i < 6; ++i) {
    linW[i] = (const float*)d_in[31 + 2 * i];
    linB[i] = (const float*)d_in[32 + 2 * i];
  }
  const float* c1w = (const float*)d_in[43];
  const float* c1b = (const float*)d_in[44];
  const float* c2w = (const float*)d_in[45];
  const float* c2b = (const float*)d_in[46];
  const float* ln1g = (const float*)d_in[47];
  const float* ln1b = (const float*)d_in[48];
  const float* ln2g = (const float*)d_in[49];
  const float* ln2b = (const float*)d_in[50];
  const float* m1w = (const float*)d_in[51];
  const float* m1b = (const float*)d_in[52];
  const float* m2w = (const float*)d_in[53];
  const float* m2b = (const float*)d_in[54];
  const float* m3w = (const float*)d_in[55];
  const float* m3b = (const float*)d_in[56];

  char* wsp = (char*)d_ws;
  size_t off = 0;
  auto alloc = [&](size_t bytes) -> void* {
    void* p = wsp + off;
    off = (off + bytes + 255) & ~(size_t)255;
    return p;
  };
  float*    hbuf   = (float*)alloc((size_t)NN * 200 * 4);
  _Float16* hf16   = (_Float16*)alloc((size_t)NN * 224 * 2);
  float*    hW     = (float*)alloc((size_t)NN * 208 * 4);
  float*    hWlin  = (float*)alloc((size_t)NN * 208 * 4);
  float*    asv    = (float*)alloc((size_t)NN * 4);
  float*    adv    = (float*)alloc((size_t)NN * 4);
  float*    h7     = (float*)alloc((size_t)NN * GCO * 4);
  _Float16* wpackA = (_Float16*)alloc((size_t)208 * 224 * 2);
  _Float16* wpackB = (_Float16*)alloc((size_t)208 * 224 * 2);
  float*    zbuf   = (float*)alloc((size_t)BGRP * 960 * 4);
  float*    znbuf  = (float*)alloc((size_t)BGRP * 960 * 4);

  static const int FI[7] = {128, 200, 200, 100, 100, 50, 50};
  static const int FO[7] = {200, 200, 100, 100, 50, 50, GCO};

#define LAUNCH_GEMM(NKT_, FOP_)                                                   \
  do {                                                                            \
    gemm_wmma<NKT_, FOP_><<<ggrid, 256, 0, stream>>>(hf16, wpackA, hW);           \
    if (!last)                                                                    \
      gemm_wmma<NKT_, FOP_><<<ggrid, 256, 0, stream>>>(hf16, wpackB, hWlin);      \
  } while (0)

  const float* Acur = x;
  for (int i = 0; i < 7; ++i) {
    int fi = FI[i], fo = FO[i];
    int fi_pad = (fi + 31) & ~31;
    int fo_pad = (fo + 15) & ~15;
    int nkt = fi_pad / 32;
    int nct = fo_pad / 16;
    int last = (i == 6);

    cast_pad<<<NN, fi_pad / 2, 0, stream>>>(Acur, hf16, fi, fi_pad);

    int rp_total = nct * nkt * 32;
    int rp_blocks = (rp_total + 255) / 256;
    repack_w<<<rp_blocks, 256, 0, stream>>>(gatW[i], wpackA, fi, fo, nkt, nct);
    if (!last)
      repack_w<<<rp_blocks, 256, 0, stream>>>(linW[i], wpackB, fi, fo, nkt, nct);

    dim3 ggrid(NN / 16 / 8, nct);
    switch (i) {
      case 0: LAUNCH_GEMM(4, 208); break;   // 128 -> 200 (fi_pad 128)
      case 1: LAUNCH_GEMM(7, 208); break;   // 200 -> 200 (fi_pad 224)
      case 2: LAUNCH_GEMM(7, 112); break;   // 200 -> 100 (fi_pad 224)
      case 3: LAUNCH_GEMM(4, 112); break;   // 100 -> 100 (fi_pad 128)
      case 4: LAUNCH_GEMM(4, 64);  break;   // 100 -> 50  (fi_pad 128)
      case 5: LAUNCH_GEMM(2, 64);  break;   // 50  -> 50  (fi_pad 64)
      default: LAUNCH_GEMM(2, 32); break;   // 50  -> 32  (fi_pad 64)
    }

    attn_dots<<<NN / 8, 256, 0, stream>>>(hW, gatAS[i], gatAD[i], asv, adv, fo, fo_pad);

    float* outp = last ? h7 : hbuf;
    int out_stride = last ? GCO : fo;
    gat_aggregate<<<NN / 8, 256, 0, stream>>>(
        esrc, hW, last ? nullptr : hWlin, asv, adv, gatB[i],
        last ? nullptr : linB[i], outp, fo, fo_pad, out_stride, last);

    Acur = hbuf;
  }
#undef LAUNCH_GEMM

  topk_gather<<<BGRP, 256, 0, stream>>>(h7, zbuf);
  layernorm_k<<<BGRP, 256, 0, stream>>>(zbuf, ln1g, ln1b, znbuf, 960);
  head_k<<<BGRP, 256, 0, stream>>>(znbuf, c1w, c1b, c2w, c2b, ln2g, ln2b,
                                   m1w, m1b, m2w, m2b, m3w, m3b, (float*)d_out);
}
